// MyLSTM_77653008712318
// MI455X (gfx1250) — matrix operand order repaired and checked by reading, not compiled, not run on hardware
//
#include <hip/hip_runtime.h>
#include <hip/hip_bf16.h>
#include <stdint.h>

// LSTM: B=32, T=512, D=1024.  out = [32,512,1024] ++ cT[32,1024] ++ hT[32,1024]
#define B_    32
#define T_    512
#define D_    1024
#define NWG   64          // each WG owns 16 hidden columns (d-slice)
#define NTHR  256         // 8 waves: wave = mt*4 + gate
// LDS: 128KB packed bf16 weights + 64KB A-stage (s(t) bf16) + 8KB z tiles
#define SMEM_BYTES (128*1024 + 64*1024 + 8*1024)

#if defined(__has_builtin)
#if __has_builtin(__builtin_amdgcn_global_load_async_to_lds_b128)
#define HAS_ASYNC_LDS 1
#endif
#endif

typedef __attribute__((ext_vector_type(16))) __bf16       v16bf;
typedef __attribute__((ext_vector_type(8)))  float        v8f;
typedef __attribute__((ext_vector_type(4)))  unsigned int v4u;
typedef __attribute__((ext_vector_type(4)))  int          v4i;
// address-space-qualified int4 pointers for the async-to-LDS builtin
typedef __attribute__((address_space(1))) v4i* gv4p;   // global (AS1)
typedef __attribute__((address_space(3))) v4i* lv4p;   // LDS (AS3)

union Frag { v4u u2[2]; v16bf bf; };

__device__ __forceinline__ uint32_t f2bf1(float f) {
  uint32_t u = __float_as_uint(f);
  uint32_t r = u + 0x7FFFu + ((u >> 16) & 1u);   // round-to-nearest-even
  return r >> 16;
}
__device__ __forceinline__ uint32_t pack2(float lo, float hi) {
  return f2bf1(lo) | (f2bf1(hi) << 16);
}
__device__ __forceinline__ float sigmoidf_(float x) {
  x = fminf(fmaxf(x, -30.f), 30.f);
  return 1.f / (1.f + __expf(-x));
}
__device__ __forceinline__ float tanhf_(float x) {
  x = fminf(fmaxf(x, -15.f), 15.f);
  float e = __expf(2.f * x);
  return (e - 1.f) / (e + 1.f);
}
__device__ __forceinline__ void wait_async0() {
#if defined(__has_builtin)
#if __has_builtin(__builtin_amdgcn_s_wait_asynccnt)
  __builtin_amdgcn_s_wait_asynccnt(0);
  return;
#else
  asm volatile("s_wait_asynccnt 0x0" ::: "memory");
  return;
#endif
#else
  asm volatile("s_wait_asynccnt 0x0" ::: "memory");
#endif
}

// ---- init: zero global barrier counter, s(0) = bf16(x[:,0,:]) -----------
__global__ void lstm_init(const float* __restrict__ x,
                          unsigned* __restrict__ ctr,
                          uint16_t* __restrict__ sbuf) {
  int idx = blockIdx.x * blockDim.x + threadIdx.x;
  if (idx == 0) ctr[0] = 0u;
  if (idx < B_ * D_) {
    int b = idx >> 10, d = idx & 1023;
    sbuf[idx] = (uint16_t)f2bf1(x[((size_t)b * T_) * D_ + d]);
  }
}

// ---- persistent LSTM kernel --------------------------------------------
__global__ void __launch_bounds__(NTHR, 1)
lstm_persist(const float* __restrict__ x,
             const float* __restrict__ W1, const float* __restrict__ b1,
             const float* __restrict__ W2, const float* __restrict__ b2,
             const float* __restrict__ W3, const float* __restrict__ b3,
             const float* __restrict__ W4, const float* __restrict__ b4,
             const float* __restrict__ bias,
             float* __restrict__ out,
             unsigned* __restrict__ ctr,
             uint16_t* __restrict__ sbuf)   // 2 x [32][1024] bf16, double buffered
{
  extern __shared__ __align__(16) char smem[];
  char*  smB = smem;                            // 128 tiles x 1KB, WMMA-B lane layout
  char*  smA = smem + 128 * 1024;               // 64KB: s(t) staged, row-major bf16
  float* smZ = (float*)(smem + 192 * 1024);     // 8 z tiles x 256 f32

  const int tid    = threadIdx.x;
  const int l      = tid & 31;
  const int w      = tid >> 5;
  const int mt     = w >> 2;        // batch-row tile: rows [mt*16, mt*16+16)
  const int g      = w & 3;         // gate: 0=f(W1) 1=i(W2) 2=g(W3) 3=o(W4)
  const int wg     = blockIdx.x;    // owns hidden cols d in [16*wg, 16*wg+16)
  const int dl     = l & 15;
  const int d      = wg * 16 + dl;
  const int hiHalf = (l >= 16) ? 1 : 0;

  // ---------- Stage weights once: gather W_g rows (contiguous in K), pack
  // bf16 pairs into a contiguous-k B lane layout (same k order as A):
  //   lanes 0-15 hold K=0..15, lanes 16-31 hold K=16..31, col N = lane&15.
  const float* Wg_[4] = {W1, W2, W3, W4};
  for (int ti = w; ti < 128; ti += 8) {          // 128 tiles = 4 gates x 32 k-tiles
    int g2 = ti >> 5, kt = ti & 31;
    int kbase = kt * 32 + (hiHalf ? 16 : 0);
    const float* src = Wg_[g2] + (size_t)d * D_ + kbase;   // Wcat[k][g*1024+d] = W_g[d][k]
    uint32_t pk[8];
#pragma unroll
    for (int j = 0; j < 8; ++j) pk[j] = pack2(src[2 * j], src[2 * j + 1]);
    v4u lo, hi;
    lo[0] = pk[0]; lo[1] = pk[1]; lo[2] = pk[2]; lo[3] = pk[3];
    hi[0] = pk[4]; hi[1] = pk[5]; hi[2] = pk[6]; hi[3] = pk[7];
    v4u* dst = (v4u*)(smB + ti * 1024 + l * 32);
    dst[0] = lo;
    dst[1] = hi;
  }

  // per-lane fused bias for this wave's gate/column
  const float* bg = (g == 0) ? b1 : (g == 1) ? b2 : (g == 2) ? b3 : b4;
  const float bv  = bg[d] + bias[d];

  float cs[8];                      // cell state, lives in VGPRs of g==0 waves
#pragma unroll
  for (int r = 0; r < 8; ++r) cs[r] = 0.f;

  __syncthreads();

  // per-lane static addresses for the GEMM loop (all LDS)
  const char* aptr  = smA + (size_t)(mt * 16 + dl) * (D_ * 2) + (hiHalf ? 32 : 0);
  const char* bptr0 = smB + (size_t)g * 32 * 1024 + l * 32;

  for (int t = 0; t < T_; ++t) {
    // ---- stage A = s(t) (64KB bf16) from global into LDS, async bulk copy ----
    {
      const char* gsrc = (const char*)(sbuf + (size_t)(t & 1) * (B_ * D_));
#ifdef HAS_ASYNC_LDS
#pragma unroll
      for (int i = 0; i < 16; ++i) {            // 4096 chunks of 16B, 16 per thread
        int c = tid + i * NTHR;
        __builtin_amdgcn_global_load_async_to_lds_b128(
            (gv4p)(gsrc + (size_t)c * 16), (lv4p)(smA + (size_t)c * 16), 0, 0);
      }
      wait_async0();
#else
#pragma unroll
      for (int i = 0; i < 16; ++i) {
        int c = tid + i * NTHR;
        *(v4u*)(smA + (size_t)c * 16) = *(const v4u*)(gsrc + (size_t)c * 16);
      }
#endif
      __syncthreads();
    }

    // ---- GEMM: z[mt-tile, gate-tile] = s(t) @ Wcat-slice, K=1024, all-LDS,
    //      register double-buffered so ds_loads for kt+1 precede wmma kt ----
    Frag Af[2], Bf[2];
    Af[0].u2[0] = *(const v4u*)(aptr + 0);
    Af[0].u2[1] = *(const v4u*)(aptr + 16);
    Bf[0].u2[0] = *(const v4u*)(bptr0 + 0);
    Bf[0].u2[1] = *(const v4u*)(bptr0 + 16);
    v8f acc = {};
#pragma unroll
    for (int kt = 0; kt < 32; ++kt) {
      const int cur = kt & 1, nxt = cur ^ 1;
      if (kt < 31) {
        const char* ap = aptr + (kt + 1) * 64;
        const char* bp = bptr0 + (kt + 1) * 1024;
        Af[nxt].u2[0] = *(const v4u*)(ap);
        Af[nxt].u2[1] = *(const v4u*)(ap + 16);
        Bf[nxt].u2[0] = *(const v4u*)(bp);
        Bf[nxt].u2[1] = *(const v4u*)(bp + 16);
      }
      acc = __builtin_amdgcn_wmma_f32_16x16x32_bf16(
          false, Af[cur].bf, false, Bf[cur].bf, (short)0, acc, false, false);
    }

    // ---- publish z tile (+bias) to LDS ----
    float* zt = smZ + w * 256;
#pragma unroll
    for (int r = 0; r < 8; ++r) zt[r * 32 + l] = acc[r] + bv;
    __syncthreads();

    // ---- gate combine on g==0 waves; c stays in registers ----
    if (g == 0) {
      const float* zf = smZ + (mt * 4 + 0) * 256;
      const float* zi = smZ + (mt * 4 + 1) * 256;
      const float* zg = smZ + (mt * 4 + 2) * 256;
      const float* zo = smZ + (mt * 4 + 3) * 256;
      uint16_t* sN = sbuf + (size_t)((t + 1) & 1) * (B_ * D_);
#pragma unroll
      for (int r = 0; r < 8; ++r) {
        int   idx = r * 32 + l;
        float f   = sigmoidf_(zf[idx]);
        float ig  = sigmoidf_(zi[idx]) * tanhf_(zg[idx]);
        float cn  = cs[r] * f + ig;
        cs[r]     = cn;
        float h   = sigmoidf_(zo[idx]) * tanhf_(cn);
        int   b   = mt * 16 + r + (hiHalf ? 8 : 0);
        out[((size_t)b * T_ + t) * D_ + d] = h;
        if (t + 1 < T_) {
          float sn = h + x[((size_t)b * T_ + (t + 1)) * D_ + d];
          sN[(size_t)b * D_ + d] = (uint16_t)f2bf1(sn);
        } else {
          out[(size_t)B_ * T_ * D_ + (size_t)b * D_ + d]                    = cn; // cT
          out[(size_t)B_ * T_ * D_ + (size_t)B_ * D_ + (size_t)b * D_ + d]  = h;  // hT
        }
      }
    }
    __syncthreads();           // z-tile/A-stage LDS reuse + s writes ordered in-WG
    __threadfence();           // release s(t+1) to device scope
    if (tid == 0) {
      __hip_atomic_fetch_add(ctr, 1u, __ATOMIC_RELEASE, __HIP_MEMORY_SCOPE_AGENT);
      unsigned tgt = (unsigned)(t + 1) * (unsigned)NWG;
      while (__hip_atomic_load(ctr, __ATOMIC_ACQUIRE, __HIP_MEMORY_SCOPE_AGENT) < tgt)
        __builtin_amdgcn_s_sleep(2);
    }
    __syncthreads();
    __threadfence();           // acquire: invalidate L0 before reading s(t+1)
  }
}

extern "C" void kernel_launch(void* const* d_in, const int* in_sizes, int n_in,
                              void* d_out, int out_size, void* d_ws, size_t ws_size,
                              hipStream_t stream) {
  (void)in_sizes; (void)n_in; (void)out_size; (void)ws_size;
  const float* x    = (const float*)d_in[0];
  const float* W1   = (const float*)d_in[1];
  const float* b1   = (const float*)d_in[2];
  const float* W2   = (const float*)d_in[3];
  const float* b2   = (const float*)d_in[4];
  const float* W3   = (const float*)d_in[5];
  const float* b3   = (const float*)d_in[6];
  const float* W4   = (const float*)d_in[7];
  const float* b4   = (const float*)d_in[8];
  const float* bias = (const float*)d_in[9];
  float*    out  = (float*)d_out;
  unsigned* ctr  = (unsigned*)d_ws;
  uint16_t* sbuf = (uint16_t*)((char*)d_ws + 256);   // 2 x 64KB bf16 s buffers

  lstm_init<<<128, 256, 0, stream>>>(x, ctr, sbuf);

  (void)hipFuncSetAttribute((const void*)lstm_persist,
                            hipFuncAttributeMaxDynamicSharedMemorySize,
                            (int)SMEM_BYTES);
  lstm_persist<<<NWG, NTHR, SMEM_BYTES, stream>>>(
      x, W1, b1, W2, b2, W3, b3, W4, b4, bias, out, ctr, sbuf);
}
